// Model_15324443312182
// MI455X (gfx1250) — matrix operand order repaired and checked by reading, not compiled, not run on hardware
//
#include <hip/hip_runtime.h>

// ---------------------------------------------------------------------------
// MI455X (gfx1250) sparse-coding pipeline:
//   1) conv_topk : im2col GEMM via v_wmma_f32_16x16x32_bf16 (split bf16 hi/lo
//                  for ~fp32 accuracy), atoms staged with
//                  global_load_async_to_lds_b128 (double-buffered, ASYNCcnt),
//                  + exact-superset top-k tile filter
//   2) filter_pass (x2) : pivot-filter candidate lists down to ~4k/batch
//   3) final_topk : exact top-64 per batch
//   4) scatter_add: out[t0+k] += value * atom[k]  (global_atomic_add_f32)
// Workspace layout (needs ~18.1 MiB):
//   [0)      counters: u32[16]   (c1=0..3, c2=4..7, c3=8..11)
//   [256)    finals  : u64[256]  (64 per batch)
//   [4096)   L1      : u64[4*CAP1], then L2 : u64[4*CAP2], then L3 : u64[4*CAP3]
// ---------------------------------------------------------------------------

#define NS   32768      // N_SAMPLES
#define NA   2048       // N_ATOMS
#define ASZ  1024       // ATOM_SIZE
#define NB   4          // BATCH
#define MT   128        // workgroup M tile (atoms)
#define NT   128        // workgroup N tile (time)
#define KCH  32         // K chunk per WMMA step
#define LCAND 2048
#define CAP1 (1u << 19)
#define CAP2 (1u << 16)
#define CAP3 4096u

#if defined(__has_builtin)
#if __has_builtin(__builtin_amdgcn_global_load_async_to_lds_b128) && \
    __has_builtin(__builtin_amdgcn_s_wait_asynccnt)
#define USE_ASYNC 1
#endif
#endif
#ifndef USE_ASYNC
#define USE_ASYNC 0
#endif

typedef __attribute__((ext_vector_type(8)))  float  v8f;
typedef __attribute__((ext_vector_type(8)))  __bf16 v8bf;
typedef __attribute__((ext_vector_type(16))) __bf16 v16bf;

// gcc-style int4 to match the async builtin's V4i pointer params
typedef int vsi4 __attribute__((vector_size(16)));
typedef __attribute__((address_space(1))) vsi4* gv4p;   // global
typedef __attribute__((address_space(3))) vsi4* lv4p;   // LDS

// ---- order-preserving float <-> uint key ----------------------------------
__device__ __forceinline__ unsigned fkey(float v) {
  unsigned u = __float_as_uint(v);
  return (u & 0x80000000u) ? ~u : (u | 0x80000000u);
}
__device__ __forceinline__ float unkey(unsigned k) {
  unsigned u = (k & 0x80000000u) ? (k ^ 0x80000000u) : ~k;
  return __uint_as_float(u);
}

// ---- 64-bit shuffle built from two 32-bit shuffles (wave32) ---------------
__device__ __forceinline__ unsigned long long sx64(unsigned long long x, int m) {
  int lo = __shfl_xor((int)(unsigned)(x & 0xFFFFFFFFull), m, 32);
  int hi = __shfl_xor((int)(unsigned)(x >> 32), m, 32);
  return ((unsigned long long)(unsigned)hi << 32) | (unsigned)lo;
}

// ---- in-wave bitonic sorts (descending), wave32 ---------------------------
__device__ __forceinline__ float waveSortDescF(float v, int lane) {
  for (int k = 2; k <= 32; k <<= 1)
    for (int j = k >> 1; j > 0; j >>= 1) {
      float p = __shfl_xor(v, j, 32);
      bool up = ((lane & k) != 0);
      bool lower = ((lane & j) == 0);
      v = (up == lower) ? fminf(v, p) : fmaxf(v, p);
    }
  return v;
}
__device__ __forceinline__ unsigned long long waveSortDesc64(unsigned long long v, int lane) {
  for (int k = 2; k <= 32; k <<= 1)
    for (int j = k >> 1; j > 0; j >>= 1) {
      unsigned long long p = sx64(v, j);
      bool up = ((lane & k) != 0);
      bool lower = ((lane & j) == 0);
      v = (up == lower) ? (v < p ? v : p) : (v > p ? v : p);
    }
  return v;
}

// ---- assemble a 16-element bf16 fragment from two aligned LDS b128 loads --
__device__ __forceinline__ v16bf ldfrag(const __bf16* p, int stride) {
  v8bf x = *(const v8bf*)p;
  v8bf y = *(const v8bf*)(p + stride);
  return __builtin_shufflevector(x, y, 0, 1, 2, 3, 4, 5, 6, 7,
                                 8, 9, 10, 11, 12, 13, 14, 15);
}

#if USE_ASYNC
// async global(16B) -> LDS copy per lane (ASYNCcnt-tracked).
// Low 32 bits of a generic LDS address are the LDS byte offset (ISA 10.2);
// AS pointers built via uintptr_t round-trip (no addrspacecast needed).
__device__ __forceinline__ void async_cp16(const float* g, float* l) {
  __builtin_amdgcn_global_load_async_to_lds_b128(
      (gv4p)(uintptr_t)g, (lv4p)(uintptr_t)l, 0, 0);
}
#endif

// ===========================================================================
// Pass 1: WMMA conv tile + pivot-filtered candidate emission
// grid (NS/NT, NA/MT, NB), block 256 (8 wave32)
// ===========================================================================
__global__ void __launch_bounds__(256)
conv_topk(const float* __restrict__ orig, const float* __restrict__ atoms,
          unsigned long long* __restrict__ L1, unsigned* __restrict__ c1) {
  const int tid  = threadIdx.x;
  const int lane = tid & 31;
  const int wid  = tid >> 5;
  const int b    = blockIdx.z;
  const int m0   = blockIdx.y * MT;
  const int t0   = blockIdx.x * NT;
  const int mW   = (wid & 3) * 32;   // 4 waves along M
  const int nW   = (wid >> 2) * 64;  // 2 waves along N
  const int row  = lane & 15;
  const int hsel = lane >> 4;

  __shared__ __align__(16) __bf16 sigH[NT + ASZ + 16];
  __shared__ __align__(16) __bf16 sigL[NT + ASZ + 16];
  __shared__ __align__(16) __bf16 AH[MT * KCH], AL[MT * KCH];
  __shared__ __align__(16) __bf16 BH[NT * KCH], BL[NT * KCH];
#if USE_ASYNC
  __shared__ __align__(16) float Araw[2][MT * KCH];   // raw f32 double buffer
#endif
  __shared__ unsigned long long lc[LCAND];
  __shared__ float w8[8];
  __shared__ unsigned lcnt, gbase;

#if USE_ASYNC
  // prologue: async-stage chunk 0 (4 b128 rounds x 256 lanes = 16 KiB)
  {
    const int e = tid * 4;
    #pragma unroll
    for (int i = 0; i < 4; ++i) {
      const int ee = e + i * 1024;
      async_cp16(&atoms[(size_t)(m0 + (ee >> 5)) * ASZ + (ee & (KCH - 1))],
                 &Araw[0][ee]);
    }
  }
#endif

  // load + hi/lo-split the padded signal window into LDS
  for (int j = tid; j < NT + ASZ; j += 256) {
    int g = t0 + j - ASZ / 2;
    float v = (g >= 0 && g < NS) ? orig[(size_t)b * NS + g] : 0.0f;
    __bf16 h = (__bf16)v;
    sigH[j] = h;
    sigL[j] = (__bf16)(v - (float)h);
  }

  v8f z8 = {0.f, 0.f, 0.f, 0.f, 0.f, 0.f, 0.f, 0.f};
  v8f acc[2][4];
#pragma unroll
  for (int mi = 0; mi < 2; ++mi)
#pragma unroll
    for (int ni = 0; ni < 4; ++ni) acc[mi][ni] = z8;

  __syncthreads();

  for (int kc = 0; kc < ASZ; kc += KCH) {
#if USE_ASYNC
    const int cur = (kc >> 5) & 1;
    __builtin_amdgcn_s_wait_asynccnt(0);   // my chunk-cur loads landed
#endif
    __syncthreads();                       // everyone's loads landed; prev WMMA reads done
#if USE_ASYNC
    if (kc + KCH < ASZ) {                  // issue chunk cur^1 while we compute
      const int e = tid * 4;
      #pragma unroll
      for (int i = 0; i < 4; ++i) {
        const int ee = e + i * 1024;
        async_cp16(&atoms[(size_t)(m0 + (ee >> 5)) * ASZ + kc + KCH + (ee & (KCH - 1))],
                   &Araw[cur ^ 1][ee]);
      }
    }
    // stage A: hi/lo split from raw f32 LDS buffer
    for (int e = tid; e < MT * KCH; e += 256) {
      float v = Araw[cur][e];
      __bf16 h = (__bf16)v;
      AH[e] = h;
      AL[e] = (__bf16)(v - (float)h);
    }
#else
    // stage A: 128x32 atoms chunk from global, split hi/lo
    for (int e = tid; e < MT * KCH; e += 256) {
      int m = e >> 5, k = e & (KCH - 1);
      float v = atoms[(size_t)(m0 + m) * ASZ + kc + k];
      __bf16 h = (__bf16)v;
      AH[e] = h;
      AL[e] = (__bf16)(v - (float)h);
    }
    if (kc + KCH < ASZ)
      __builtin_prefetch(&atoms[(size_t)(m0 + (tid >> 1)) * ASZ + kc + KCH], 0, 1);
#endif
    // stage B: 128x32 sliding-window chunk from LDS signal
    for (int e = tid; e < NT * KCH; e += 256) {
      int n = e >> 5, k = e & (KCH - 1);
      BH[e] = sigH[n + kc + k];
      BL[e] = sigL[n + kc + k];
    }
    __syncthreads();

    // A fragments (16x32 bf16): lanes<16 hold K0-7/16-23, lanes>=16 K8-15/24-31
    v16bf aH[2], aL[2];
#pragma unroll
    for (int mi = 0; mi < 2; ++mi) {
      const int off = (mW + mi * 16 + row) * KCH + hsel * 8;
      aH[mi] = ldfrag(&AH[off], 16);
      aL[mi] = ldfrag(&AL[off], 16);
    }
    // B fragments (32x16 bf16): lanes<16 hold K0-7/16-23, lanes>=16 K8-15/24-31
#pragma unroll
    for (int ni = 0; ni < 4; ++ni) {
      const int off = (nW + ni * 16 + row) * KCH + hsel * 8;
      v16bf bH = ldfrag(&BH[off], 16);
      v16bf bL = ldfrag(&BL[off], 16);
#pragma unroll
      for (int mi = 0; mi < 2; ++mi) {
        acc[mi][ni] = __builtin_amdgcn_wmma_f32_16x16x32_bf16(
            false, aH[mi], false, bH, (short)0, acc[mi][ni], false, false);
        acc[mi][ni] = __builtin_amdgcn_wmma_f32_16x16x32_bf16(
            false, aH[mi], false, bL, (short)0, acc[mi][ni], false, false);
        acc[mi][ni] = __builtin_amdgcn_wmma_f32_16x16x32_bf16(
            false, aL[mi], false, bH, (short)0, acc[mi][ni], false, false);
      }
    }
  }

  // ---- tile top-k filter: pivot tau = min over waves of 8th-largest thread-max
  float tmax = -3.4e38f;
#pragma unroll
  for (int mi = 0; mi < 2; ++mi)
#pragma unroll
    for (int ni = 0; ni < 4; ++ni)
#pragma unroll
      for (int e = 0; e < 8; ++e) tmax = fmaxf(tmax, acc[mi][ni][e]);

  float srt = waveSortDescF(tmax, lane);
  __syncthreads();   // conv stage buffers idle before selection scratch reuse
  if (lane == 7) w8[wid] = srt;
  if (tid == 0) lcnt = 0;
  __syncthreads();

  float tau = w8[0];
#pragma unroll
  for (int i = 1; i < 8; ++i) tau = fminf(tau, w8[i]);
  const unsigned tk = fkey(tau);

  const int idxBase = ((m0 + mW + hsel * 8) << 15) + (t0 + nW + row);
#pragma unroll
  for (int mi = 0; mi < 2; ++mi)
#pragma unroll
    for (int ni = 0; ni < 4; ++ni)
#pragma unroll
      for (int e = 0; e < 8; ++e) {
        unsigned k = fkey(acc[mi][ni][e]);
        if (k >= tk) {
          unsigned pos = atomicAdd(&lcnt, 1u);
          if (pos < LCAND) {
            int idx = idxBase + ((mi * 16 + e) << 15) + ni * 16;
            lc[pos] = ((unsigned long long)k << 32) | (unsigned)idx;
          }
        }
      }
  __syncthreads();

  unsigned cnt = lcnt;
  if (cnt > LCAND) cnt = LCAND;
  if (tid == 0) gbase = atomicAdd(&c1[b], cnt);
  __syncthreads();
  unsigned gb = gbase;
  for (unsigned i = tid; i < cnt; i += 256) {
    unsigned g = gb + i;
    if (g < CAP1) L1[(size_t)b * CAP1 + g] = lc[i];
  }
}

// ===========================================================================
// Generic pivot filter over packed u64 candidates; slice of 8192 per block
// grid (cap_in/8192, NB)
// ===========================================================================
__global__ void __launch_bounds__(256)
filter_pass(const unsigned long long* __restrict__ in,
            const unsigned* __restrict__ cin, unsigned cap_in,
            unsigned long long* __restrict__ outl,
            unsigned* __restrict__ cout, unsigned cap_out) {
  const int tid = threadIdx.x, lane = tid & 31, wid = tid >> 5;
  const int b = blockIdx.y;
  unsigned n = cin[b];
  if (n > cap_in) n = cap_in;
  const size_t base = (size_t)b * cap_in;
  const unsigned s0 = blockIdx.x * 8192u;

  __shared__ unsigned long long lc[CAP3];
  __shared__ unsigned long long w8[8];
  __shared__ unsigned lcnt, gbase;

  unsigned long long tmax = 0;
  for (unsigned i = s0 + tid; i < s0 + 8192u; i += 256) {
    unsigned long long v = (i < n) ? in[base + i] : 0ull;
    tmax = (v > tmax) ? v : tmax;
  }
  unsigned long long srt = waveSortDesc64(tmax, lane);
  if (lane == 7) w8[wid] = srt;
  if (tid == 0) lcnt = 0;
  __syncthreads();

  unsigned long long tau = w8[0];
#pragma unroll
  for (int i = 1; i < 8; ++i) tau = (w8[i] < tau) ? w8[i] : tau;

  for (unsigned i = s0 + tid; i < s0 + 8192u; i += 256) {
    unsigned long long v = (i < n) ? in[base + i] : 0ull;
    if (v != 0ull && v >= tau) {
      unsigned pos = atomicAdd(&lcnt, 1u);
      if (pos < CAP3) lc[pos] = v;
    }
  }
  __syncthreads();
  unsigned cnt = lcnt;
  if (cnt > CAP3) cnt = CAP3;
  if (tid == 0) gbase = atomicAdd(&cout[b], cnt);
  __syncthreads();
  unsigned gb = gbase;
  for (unsigned i = tid; i < cnt; i += 256) {
    unsigned g = gb + i;
    if (g < cap_out) outl[(size_t)b * cap_out + g] = lc[i];
  }
}

// ===========================================================================
// Exact top-64 per batch from <= CAP3 candidates; grid (NB)
// ===========================================================================
__global__ void __launch_bounds__(256)
final_topk(const unsigned long long* __restrict__ in,
           const unsigned* __restrict__ cin,
           unsigned long long* __restrict__ fin) {
  const int tid = threadIdx.x, lane = tid & 31, wid = tid >> 5;
  const int b = blockIdx.x;
  unsigned n = cin[b];
  if (n > CAP3) n = CAP3;

  __shared__ unsigned long long buf[CAP3];
  __shared__ unsigned long long red[8];

  for (unsigned i = tid; i < CAP3; i += 256)
    buf[i] = (i < n) ? in[(size_t)b * CAP3 + i] : 0ull;
  __syncthreads();

  for (int it = 0; it < 64; ++it) {
    unsigned long long m = 0;
    int arg = -1;
    for (int i = tid; i < (int)CAP3; i += 256) {
      unsigned long long v = buf[i];
      if (v > m) { m = v; arg = i; }
    }
    unsigned long long wm = m;
#pragma unroll
    for (int j = 16; j > 0; j >>= 1) {
      unsigned long long p = sx64(wm, j);
      wm = (p > wm) ? p : wm;
    }
    if (lane == 0) red[wid] = wm;
    __syncthreads();
    unsigned long long g = red[0];
#pragma unroll
    for (int i = 1; i < 8; ++i) g = (red[i] > g) ? red[i] : g;
    if (m == g && m != 0ull && arg >= 0) buf[arg] = 0ull;  // unique winner
    if (tid == 0) fin[b * 64 + it] = g;
    __syncthreads();
  }
}

// ===========================================================================
// Scatter: out[b, t+k] += atoms[a,k] * value  (crop at NS); grid (64, NB)
// ===========================================================================
__global__ void __launch_bounds__(256)
scatter_add(const unsigned long long* __restrict__ fin,
            const float* __restrict__ atoms, float* __restrict__ out) {
  const int b = blockIdx.y, j = blockIdx.x;
  unsigned long long p = fin[b * 64 + j];
  if (p == 0ull) return;
  float val = unkey((unsigned)(p >> 32));
  int idx = (int)(p & 0xFFFFFFFFull);
  int a = idx >> 15;
  int t = idx & (NS - 1);
  for (int k = threadIdx.x; k < ASZ; k += 256) {
    int pos = t + k;
    if (pos < NS)
      atomicAdd(&out[(size_t)b * NS + pos], atoms[(size_t)a * ASZ + k] * val);
  }
}

__global__ void __launch_bounds__(256)
init_kernel(float* __restrict__ out, unsigned* __restrict__ counters,
            unsigned long long* __restrict__ fin) {
  int i = blockIdx.x * 256 + threadIdx.x;
  if (i < NB * NS) out[i] = 0.0f;
  if (i < 16) counters[i] = 0u;
  if (i < NB * 64) fin[i] = 0ull;
}

// ===========================================================================
extern "C" void kernel_launch(void* const* d_in, const int* in_sizes, int n_in,
                              void* d_out, int out_size, void* d_ws, size_t ws_size,
                              hipStream_t stream) {
  const float* orig  = (const float*)d_in[0];   // (4,1,32768) f32
  const float* atoms = (const float*)d_in[1];   // (2048,1024) f32
  float* out = (float*)d_out;                   // (4,1,32768) f32

  char* ws = (char*)d_ws;
  unsigned* counters          = (unsigned*)ws;                         // 16 u32
  unsigned long long* fin     = (unsigned long long*)(ws + 256);       // 256 u64
  unsigned long long* L1      = (unsigned long long*)(ws + 4096);
  unsigned long long* L2      = L1 + (size_t)NB * CAP1;
  unsigned long long* L3      = L2 + (size_t)NB * CAP2;

  init_kernel<<<(NB * NS + 255) / 256, 256, 0, stream>>>(out, counters, fin);

  conv_topk<<<dim3(NS / NT, NA / MT, NB), 256, 0, stream>>>(orig, atoms, L1, counters);

  filter_pass<<<dim3(CAP1 / 8192, NB), 256, 0, stream>>>(L1, counters, CAP1,
                                                         L2, counters + 4, CAP2);
  filter_pass<<<dim3(CAP2 / 8192, NB), 256, 0, stream>>>(L2, counters + 4, CAP2,
                                                         L3, counters + 8, CAP3);
  final_topk<<<NB, 256, 0, stream>>>(L3, counters + 8, fin);

  scatter_add<<<dim3(64, NB), 256, 0, stream>>>(fin, atoms, out);
}
